// IDGNN_22574348108104
// MI455X (gfx1250) — matrix operand order repaired and checked by reading, not compiled, not run on hardware
//
#include <hip/hip_runtime.h>
#include <stdint.h>

#define Nn  512
#define Dm  128
#define Ee  8192

typedef __attribute__((ext_vector_type(16))) _Float16 v16h;
typedef __attribute__((ext_vector_type(8)))  _Float16 v8h;
typedef __attribute__((ext_vector_type(8)))  float    v8f;

struct MlpP { const float *W1, *b1, *W2, *b2; };
struct AllP { MlpP p0[3]; MlpP p1[3]; };

// ---------------- prep kernels: adjacency + 2-hop mask as bitrows ----------------

__global__ void idgnn_zero_bits(uint32_t* bits, int nwords) {
    int idx = blockIdx.x * blockDim.x + threadIdx.x;
    if (idx < nwords) bits[idx] = 0u;
}

__global__ void idgnn_scatter_edges(const int* __restrict__ ei, uint32_t* __restrict__ Abits) {
    int e = blockIdx.x * blockDim.x + threadIdx.x;
    if (e < Ee) {
        int r = ei[e];        // edge_index[0][e]
        int c = ei[Ee + e];   // edge_index[1][e]
        atomicOr((unsigned int*)&Abits[r * 16 + (c >> 5)], 1u << (c & 31));
    }
}

// kbits[i] = OR over m with A[i][m]==1 of Abits[m]   (== row i of (A@A)>0)
__global__ void idgnn_kmask(const uint32_t* __restrict__ Abits, uint32_t* __restrict__ kbits) {
    __shared__ uint32_t arow[16];
    const int i = blockIdx.x;
    const int t = threadIdx.x;
    if (t < 16) arow[t] = Abits[i * 16 + t];
    __syncthreads();
    if (t < 16) {
        uint32_t acc = 0u;
        for (int m = 0; m < Nn; ++m)
            if ((arow[m >> 5] >> (m & 31)) & 1u) acc |= Abits[m * 16 + t];
        kbits[i * 16 + t] = acc;
    }
}

// ---------------- WMMA GEMM: D(512x128) = src(512x128) @ W(128x128) + bias ----------------
// src fp16 row-major in LDS; Wt = W^T fp16 in LDS ([n][k]); fp32 accumulate; optional relu.
// Each wave owns one fixed 16-col N-tile (wave id == N-tile id, 8 waves == 128 cols),
// preloads its 4 B K-fragments into registers, and streams M-tiles two at a time so the
// two accumulator chains' WMMAs interleave (fills the F16 WMMA->WMMA hazard slots).
__device__ __forceinline__ void gemm128(const _Float16* __restrict__ Ab,
                                        const _Float16* __restrict__ Wt,
                                        const float* __restrict__ bias,
                                        _Float16* __restrict__ Dst,
                                        int lane, int wave, bool do_relu)
{
    const int r  = lane & 15;   // row-in-tile (A) / col-in-tile (B,C,D)
    const int hi = lane >> 4;   // lane-half selector per ISA 7.12.2
    const int n0 = wave << 4;   // this wave's N-tile

    // B 32x16 fp16 fragments: lane holds column n0+r, K = k0 + hi*16 + (0..15)
    v16h bfrag[4];
    #pragma unroll
    for (int kk = 0; kk < 4; ++kk) {
        const _Float16* bp = Wt + (n0 + r) * Dm + kk * 32 + hi * 16;
        v8h b0 = *(const v8h*)(bp);
        v8h b1 = *(const v8h*)(bp + 8);
        #pragma unroll
        for (int e = 0; e < 8; ++e) { bfrag[kk][e] = b0[e]; bfrag[kk][e + 8] = b1[e]; }
    }
    const float bv = bias[n0 + r];

    for (int mt = 0; mt < Nn / 16; mt += 2) {
        const int m0 = mt << 4;
        v8f acc0, acc1;
        #pragma unroll
        for (int j = 0; j < 8; ++j) { acc0[j] = bv; acc1[j] = bv; }   // fold bias into C
        #pragma unroll
        for (int kk = 0; kk < 4; ++kk) {
            const int k0 = kk * 32;
            // A 16x32 fp16: lanes0-15 row M, K=k0+0..15; lanes16-31 K=k0+16..31
            const _Float16* ap0 = Ab + (m0 + r) * Dm + k0 + hi * 16;
            const _Float16* ap1 = ap0 + 16 * Dm;
            v8h x0 = *(const v8h*)(ap0);
            v8h x1 = *(const v8h*)(ap0 + 8);
            v8h y0 = *(const v8h*)(ap1);
            v8h y1 = *(const v8h*)(ap1 + 8);
            v16h av0, av1;
            #pragma unroll
            for (int e = 0; e < 8; ++e) {
                av0[e] = x0[e]; av0[e + 8] = x1[e];
                av1[e] = y0[e]; av1[e + 8] = y1[e];
            }
            acc0 = __builtin_amdgcn_wmma_f32_16x16x32_f16(
                false, av0, false, bfrag[kk], (short)0, acc0, false, false);
            acc1 = __builtin_amdgcn_wmma_f32_16x16x32_f16(
                false, av1, false, bfrag[kk], (short)0, acc1, false, false);
        }
        // D: lane -> col n0+r, VGPR j -> row m0 + hi*8 + j
        _Float16* dp0 = Dst + (m0 + hi * 8) * Dm + n0 + r;
        _Float16* dp1 = dp0 + 16 * Dm;
        #pragma unroll
        for (int j = 0; j < 8; ++j) {
            float v0 = acc0[j];
            float v1 = acc1[j];
            if (do_relu) { v0 = v0 > 0.f ? v0 : 0.f; v1 = v1 > 0.f ? v1 : 0.f; }
            dp0[j * Dm] = (_Float16)v0;
            dp1[j * Dm] = (_Float16)v1;
        }
    }
}

// ---------------- main: one workgroup per target node i ----------------

__global__ void __launch_bounds__(256)
idgnn_main(const float* __restrict__ x,
           const uint32_t* __restrict__ Abits,
           const uint32_t* __restrict__ kbits,
           AllP P,
           float* __restrict__ out)
{
    extern __shared__ char smem[];
    _Float16* bufA = (_Float16*)smem;              // 512x128 fp16 (128 KB)
    _Float16* bufB = bufA + Nn * Dm;               // 512x128 fp16 (128 KB)
    _Float16* Wt   = bufB + Nn * Dm;               // 128x128 fp16 (32 KB), transposed
    float* hiRow = (float*)(Wt + Dm * Dm);         // 128 f32
    float* tmp1  = hiRow + Dm;                     // 128 f32
    float* h1v   = tmp1 + Dm;                      // 128 f32
    float* biasL = h1v + Dm;                       // 128 f32

    const int i    = blockIdx.x;
    const int tid  = threadIdx.x;
    const int lane = tid & 31;
    const int wave = tid >> 5;

    // h <- x (fp16)
    for (int idx = tid; idx < Nn * Dm; idx += 256) bufA[idx] = (_Float16)x[idx];
    __syncthreads();

    _Float16* h = bufA;   // current hidden state; becomes h0 after 2nd GEMM
    _Float16* t = bufB;   // relu intermediate; becomes h_next after aggregation

    #pragma unroll
    for (int l = 0; l < 3; ++l) {
        const MlpP p0 = P.p0[l];
        const MlpP p1 = P.p1[l];

        __builtin_prefetch(p0.W1, 0, 1);
        __builtin_prefetch(p0.W2, 0, 1);

        // save h[i] (input to identity MLP) before h gets overwritten by h0
        if (tid < Dm) hiRow[tid] = (float)h[i * Dm + tid];
        __syncthreads();

        // identity MLP p1 on single row, fp32, weights straight from global (L2)
        if (tid < Dm) {
            float acc = p1.b1[tid];
            for (int k = 0; k < Dm; ++k) acc += hiRow[k] * p1.W1[k * Dm + tid];
            tmp1[tid] = acc > 0.f ? acc : 0.f;
        }
        __syncthreads();
        if (tid < Dm) {
            float acc = p1.b2[tid];
            for (int k = 0; k < Dm; ++k) acc += tmp1[k] * p1.W2[k * Dm + tid];
            h1v[tid] = acc;
        }

        // stage W1^T + b1, then t = relu(h @ W1 + b1)
        for (int idx = tid; idx < Dm * Dm; idx += 256) {
            int kk = idx >> 7, nn2 = idx & 127;
            Wt[nn2 * Dm + kk] = (_Float16)p0.W1[idx];
        }
        if (tid < Dm) biasL[tid] = p0.b1[tid];
        __syncthreads();
        gemm128(h, Wt, biasL, t, lane, wave, true);
        __syncthreads();

        // stage W2^T + b2, then h0 = t @ W2 + b2 (overwrites h in place)
        for (int idx = tid; idx < Dm * Dm; idx += 256) {
            int kk = idx >> 7, nn2 = idx & 127;
            Wt[nn2 * Dm + kk] = (_Float16)p0.W2[idx];
        }
        if (tid < Dm) biasL[tid] = p0.b2[tid];
        __syncthreads();
        gemm128(t, Wt, biasL, h, lane, wave, false);   // h now holds h0
        __syncthreads();

        // aggregation: h_next[n] = (km[i,n] ? A@h0 + A[n,i]*(h1-h0[i]) : h0[n]) + h0[n]
        const int f0 = lane * 4;
        if (l < 2) {
            for (int n = wave * 64; n < wave * 64 + 64; ++n) {
                const uint32_t* arow = Abits + n * 16;
                float a0 = 0.f, a1 = 0.f, a2 = 0.f, a3 = 0.f;
                #pragma unroll 1
                for (int w = 0; w < 16; ++w) {
                    uint32_t bits = arow[w];
                    while (bits) {
                        int b = __builtin_ctz(bits); bits &= bits - 1u;
                        const _Float16* hp = h + (w * 32 + b) * Dm + f0;
                        a0 += (float)hp[0]; a1 += (float)hp[1];
                        a2 += (float)hp[2]; a3 += (float)hp[3];
                    }
                }
                if ((arow[i >> 5] >> (i & 31)) & 1u) {   // neighbor m == i: swap in identity row
                    const _Float16* hp = h + i * Dm + f0;
                    a0 += h1v[f0 + 0] - (float)hp[0]; a1 += h1v[f0 + 1] - (float)hp[1];
                    a2 += h1v[f0 + 2] - (float)hp[2]; a3 += h1v[f0 + 3] - (float)hp[3];
                }
                const bool km = (kbits[i * 16 + (n >> 5)] >> (n & 31)) & 1u;
                const _Float16* hn = h + n * Dm + f0;
                float b0 = (float)hn[0], b1 = (float)hn[1], b2 = (float)hn[2], b3 = (float)hn[3];
                _Float16* tp = t + n * Dm + f0;
                tp[0] = (_Float16)((km ? a0 : b0) + b0);
                tp[1] = (_Float16)((km ? a1 : b1) + b1);
                tp[2] = (_Float16)((km ? a2 : b2) + b2);
                tp[3] = (_Float16)((km ? a3 : b3) + b3);
            }
        } else if (wave == 0) {
            // final layer: only row i of h_next is the output
            const int n = i;
            const uint32_t* arow = Abits + n * 16;
            float a0 = 0.f, a1 = 0.f, a2 = 0.f, a3 = 0.f;
            #pragma unroll 1
            for (int w = 0; w < 16; ++w) {
                uint32_t bits = arow[w];
                while (bits) {
                    int b = __builtin_ctz(bits); bits &= bits - 1u;
                    const _Float16* hp = h + (w * 32 + b) * Dm + f0;
                    a0 += (float)hp[0]; a1 += (float)hp[1];
                    a2 += (float)hp[2]; a3 += (float)hp[3];
                }
            }
            if ((arow[i >> 5] >> (i & 31)) & 1u) {
                const _Float16* hp = h + i * Dm + f0;
                a0 += h1v[f0 + 0] - (float)hp[0]; a1 += h1v[f0 + 1] - (float)hp[1];
                a2 += h1v[f0 + 2] - (float)hp[2]; a3 += h1v[f0 + 3] - (float)hp[3];
            }
            const bool km = (kbits[i * 16 + (n >> 5)] >> (n & 31)) & 1u;
            const _Float16* hn = h + n * Dm + f0;
            float b0 = (float)hn[0], b1 = (float)hn[1], b2 = (float)hn[2], b3 = (float)hn[3];
            out[i * Dm + f0 + 0] = (km ? a0 : b0) + b0;
            out[i * Dm + f0 + 1] = (km ? a1 : b1) + b1;
            out[i * Dm + f0 + 2] = (km ? a2 : b2) + b2;
            out[i * Dm + f0 + 3] = (km ? a3 : b3) + b3;
        }
        __syncthreads();

        // ping-pong: h_next lives in t
        _Float16* sw = h; h = t; t = sw;
    }
}

// ---------------- host launcher ----------------

extern "C" void kernel_launch(void* const* d_in, const int* in_sizes, int n_in,
                              void* d_out, int out_size, void* d_ws, size_t ws_size,
                              hipStream_t stream) {
    (void)in_sizes; (void)n_in; (void)out_size; (void)ws_size;

    const float* x  = (const float*)d_in[0];
    const int*   ei = (const int*)d_in[1];

    AllP P;
    for (int l = 0; l < 3; ++l) {
        const int base = 2 + l * 8;
        P.p0[l].W1 = (const float*)d_in[base + 0];
        P.p0[l].b1 = (const float*)d_in[base + 1];
        P.p0[l].W2 = (const float*)d_in[base + 2];
        P.p0[l].b2 = (const float*)d_in[base + 3];
        P.p1[l].W1 = (const float*)d_in[base + 4];
        P.p1[l].b1 = (const float*)d_in[base + 5];
        P.p1[l].W2 = (const float*)d_in[base + 6];
        P.p1[l].b2 = (const float*)d_in[base + 7];
    }

    uint32_t* Abits = (uint32_t*)d_ws;            // 512*16 words = 32 KB
    uint32_t* kbits = Abits + Nn * 16;            // 512*16 words = 32 KB

    idgnn_zero_bits<<<(2 * Nn * 16 + 255) / 256, 256, 0, stream>>>(Abits, 2 * Nn * 16);
    idgnn_scatter_edges<<<(Ee + 255) / 256, 256, 0, stream>>>(ei, Abits);
    idgnn_kmask<<<Nn, 64, 0, stream>>>(Abits, kbits);

    // LDS: 2*512*128*2 + 128*128*2 + 4*128*4 = 296960 B (< 320 KB WGP limit)
    const int SMEM_BYTES = 2 * Nn * Dm * 2 + Dm * Dm * 2 + 4 * Dm * 4;
    hipFuncSetAttribute((const void*)idgnn_main,
                        hipFuncAttributeMaxDynamicSharedMemorySize, SMEM_BYTES);
    idgnn_main<<<Nn, 256, SMEM_BYTES, stream>>>(x, Abits, kbits, P, (float*)d_out);
}
